// QAttention_73967926772072
// MI455X (gfx1250) — 8-bit matrix operand order repaired and checked by reading, compiled, not run on hardware
//
#include <hip/hip_runtime.h>
#include <cstdint>


#define BATCH   16
#define SEQ     577
#define NPAD    592
#define KPAD    640
#define CTILES  37
#define CDIM    768
#define HEADS   12
#define HD      64
#define BH      (BATCH*HEADS)
#define MROWS   (BATCH*SEQ)
#define MTILES  (MROWS/16)

typedef __attribute__((ext_vector_type(8))) int v8i;


#if defined(__has_builtin)
#if __has_builtin(__builtin_amdgcn_tensor_load_to_lds) && \
    __has_builtin(__builtin_amdgcn_s_wait_tensorcnt)
#define HAVE_TDM 1
#endif
#endif

#ifdef HAVE_TDM
typedef __attribute__((ext_vector_type(4))) unsigned int tdm_v4u;
typedef __attribute__((ext_vector_type(8))) int          tdm_v8i;
typedef __attribute__((ext_vector_type(4))) int          tdm_v4i;

__device__ __forceinline__ void tdm_load_1d(const void* gptr, void* lptr,
                                            unsigned bytes) {
    unsigned long long ga = (unsigned long long)gptr;
    unsigned lds = (unsigned)(unsigned long long)lptr;
    unsigned n8 = bytes >> 3;

    tdm_v4u g0;
    g0[0] = 1u;
    g0[1] = lds;
    g0[2] = (unsigned)ga;
    g0[3] = (unsigned)((ga >> 32) & 0x01FFFFFFu) | (2u << 30);

    tdm_v8i g1;
    g1[0] = (int)(3u << 16);
    g1[1] = (int)((n8 & 0xFFFFu) << 16);
    g1[2] = (int)(((n8 >> 16) & 0xFFFFu)
                  | (1u << 16));
    g1[3] = (int)((n8 & 0xFFFFu) << 16);
    g1[4] = 0;
    g1[5] = (int)n8;
    g1[6] = 0;
    g1[7] = 0;

    tdm_v4i z4 = {0, 0, 0, 0};
#if __clang_major__ >= 23
    tdm_v8i z8 = {0, 0, 0, 0, 0, 0, 0, 0};
    __builtin_amdgcn_tensor_load_to_lds(g0, g1, z4, z4, z8, 0);
#else
    __builtin_amdgcn_tensor_load_to_lds(g0, g1, z4, z4, 0);
#endif
}
#endif

__device__ __forceinline__ void lds_strip_load(int8_t* lds, const int8_t* g,
                                               unsigned bytes) {
#ifdef HAVE_TDM
    tdm_load_1d(g, lds, bytes);
    __builtin_amdgcn_s_wait_tensorcnt(0);
#else
    int lane = threadIdx.x & 31;
    for (unsigned i = (unsigned)lane * 16; i < bytes; i += 32u * 16u)
        *(int4*)(lds + i) = *(const int4*)(g + i);
#endif
}


__device__ __forceinline__ v8i wmma_zero() {
    v8i z = {0,0,0,0,0,0,0,0};
    return z;
}

__device__ __forceinline__ v8i wmma_load_a_i8(const int8_t* a, int lda) {
    int lane = threadIdx.x & 31;
    int m = lane & 15, half = lane >> 4;
    const int8_t* p = a + (long long)m * lda + half * 8;
    int2 c0 = *(const int2*)(p +  0);
    int2 c1 = *(const int2*)(p + 16);
    int2 c2 = *(const int2*)(p + 32);
    int2 c3 = *(const int2*)(p + 48);
    v8i r;
    r[0]=c0.x; r[1]=c0.y; r[2]=c1.x; r[3]=c1.y;
    r[4]=c2.x; r[5]=c2.y; r[6]=c3.x; r[7]=c3.y;
    return r;
}

__device__ __forceinline__ v8i wmma_load_b_i8(const int8_t* b, int ldb) {
    int lane = threadIdx.x & 31;
    int n = lane & 15, half = lane >> 4;
    const int8_t* p = b + (long long)n * ldb + half * 8;
    int2 c0 = *(const int2*)(p +  0);
    int2 c1 = *(const int2*)(p + 16);
    int2 c2 = *(const int2*)(p + 32);
    int2 c3 = *(const int2*)(p + 48);
    v8i r;
    r[0]=c0.x; r[1]=c0.y; r[2]=c1.x; r[3]=c1.y;
    r[4]=c2.x; r[5]=c2.y; r[6]=c3.x; r[7]=c3.y;
    return r;
}

__device__ __forceinline__ v8i wmma_iu8(v8i a, v8i b, v8i c) {
    return __builtin_amdgcn_wmma_i32_16x16x64_iu8(true, a, true, b, c, false, false);
}


__device__ __forceinline__ float mkscale(float amax) {
    return fmaxf(amax * (1.0f / 127.0f), 1e-8f);
}

__device__ __forceinline__ int8_t q8(float x, float inv_s) {
    float r = rintf(x * inv_s);
    r = fminf(fmaxf(r, -127.0f), 127.0f);
    return (int8_t)r;
}

__device__ __forceinline__ float wave_max32(float v) {
    #pragma unroll
    for (int off = 16; off; off >>= 1)
        v = fmaxf(v, __shfl_xor(v, off, 32));
    return v;
}


__global__ void init_scalars_kernel(unsigned* s) {
    int t = threadIdx.x;
    if (t < 7) s[t] = 0u;
    if (t == 7) s[7] = 0x7F800000u;
}

__global__ void absmax_kernel(const float* __restrict__ x, long long n,
                              unsigned* out) {
    long long i = (long long)blockIdx.x * blockDim.x + threadIdx.x;
    long long stride = (long long)gridDim.x * blockDim.x;
    float m = 0.0f;
    for (; i < n; i += stride) m = fmaxf(m, fabsf(x[i]));
    m = wave_max32(m);
    if ((threadIdx.x & 31) == 0) atomicMax(out, __float_as_uint(m));
}

__global__ void quantize_kernel(const float* __restrict__ x,
                                int8_t* __restrict__ q, long long n,
                                const float* __restrict__ amax) {
    float inv = 1.0f / mkscale(amax[0]);
    long long i = (long long)blockIdx.x * blockDim.x + threadIdx.x;
    long long stride = (long long)gridDim.x * blockDim.x;
    for (; i < n; i += stride) q[i] = q8(x[i], inv);
}

__global__ void quantize_qk_kernel(const float* __restrict__ src,
                                   int8_t* __restrict__ dst,
                                   const float* __restrict__ amax) {
    const long long n = (long long)BH * NPAD * HD;
    float inv = 1.0f / mkscale(amax[0]);
    long long i = (long long)blockIdx.x * blockDim.x + threadIdx.x;
    long long stride = (long long)gridDim.x * blockDim.x;
    for (; i < n; i += stride) {
        int row = (int)((i >> 6) % NPAD);
        int8_t v = 0;
        if (row < SEQ) v = q8(src[i], inv);
        dst[i] = v;
    }
}

__global__ void quantize_vT_kernel(const float* __restrict__ src,
                                   int8_t* __restrict__ dst,
                                   const float* __restrict__ amax) {
    const long long n = (long long)BH * HD * KPAD;
    float inv = 1.0f / mkscale(amax[0]);
    long long i = (long long)blockIdx.x * blockDim.x + threadIdx.x;
    long long stride = (long long)gridDim.x * blockDim.x;
    for (; i < n; i += stride) {
        long long bh = i / (HD * KPAD);
        int rem = (int)(i % (HD * KPAD));
        int d = rem / KPAD, m = rem % KPAD;
        int8_t v = 0;
        if (m < SEQ)
            v = q8(src[bh * NPAD * HD + (long long)m * HD + d], inv);
        dst[i] = v;
    }
}

__global__ void gemm_qkv_kernel(const int8_t* __restrict__ xq,
                                const int8_t* __restrict__ wq,
                                const float*  __restrict__ bias,
                                const float*  __restrict__ scal,
                                unsigned*     __restrict__ amax3,
                                float* __restrict__ qf,
                                float* __restrict__ kf,
                                float* __restrict__ vf) {
    __shared__ int8_t Ws[64 * CDIM];

    int g  = blockIdx.x;
    int wv = threadIdx.x >> 5;
    int mtile = blockIdx.y * 4 + wv;
    int lane = threadIdx.x & 31, half = lane >> 4, nn = lane & 15;

    const int8_t* bstrip = wq + (long long)g * 64 * CDIM;
    lds_strip_load(Ws + wv * 16 * CDIM, bstrip + (long long)wv * 16 * CDIM,
                   16 * CDIM);
    __syncthreads();

    bool valid = (mtile < MTILES);
    int  mt    = valid ? mtile : (MTILES - 1);

    float sxw = mkscale(scal[0]) * mkscale(scal[1]);
    const int8_t* abase = xq + (long long)mt * 16 * CDIM;

    v8i acc[4] = { wmma_zero(), wmma_zero(), wmma_zero(), wmma_zero() };
    for (int kb = 0; kb < CDIM; kb += 64) {
        v8i a = wmma_load_a_i8(abase + kb, CDIM);
        #pragma unroll
        for (int nt = 0; nt < 4; ++nt) {
            v8i b = wmma_load_b_i8(Ws + nt * 16 * CDIM + kb, CDIM);
            acc[nt] = wmma_iu8(a, b, acc[nt]);
        }
    }

    if (valid) {
        int t = g / 12;
        int obase = g * 64 - t * CDIM;
        float* dst = (t == 0) ? qf : (t == 1) ? kf : vf;
        float post = (t == 0) ? 0.125f : 1.0f;

        float lmax = 0.0f;
        #pragma unroll
        for (int nt = 0; nt < 4; ++nt) {
            int oc = obase + nt * 16 + nn;
            int h = oc >> 6, d = oc & 63;
            float bv = bias[g * 64 + nt * 16 + nn];
            #pragma unroll
            for (int r = 0; r < 8; ++r) {
                int mrow = mt * 16 + half * 8 + r;
                int b_ = mrow / SEQ, nrow = mrow % SEQ;
                float v = ((float)acc[nt][r] * sxw + bv) * post;
                dst[(((long long)b_ * HEADS + h) * NPAD + nrow) * HD + d] = v;
                lmax = fmaxf(lmax, fabsf(v));
            }
        }
        lmax = wave_max32(lmax);
        if (lane == 0) atomicMax(amax3 + t, __float_as_uint(lmax));
    }
}

__global__ void attn_pass1_kernel(const int8_t* __restrict__ qq,
                                  const int8_t* __restrict__ kq,
                                  const float*  __restrict__ scal,
                                  unsigned*     __restrict__ zmin,
                                  float* __restrict__ rowmax,
                                  float* __restrict__ rowsum) {
    __shared__ float S[16][NPAD];
    int rt = blockIdx.x, bh = blockIdx.y;
    int lane = threadIdx.x & 31, half = lane >> 4, nn = lane & 15;
    float sqk = mkscale(scal[3]) * mkscale(scal[4]);

    const int8_t* qbase = qq + (long long)bh * NPAD * HD + (long long)rt * 16 * HD;
    const int8_t* kbase = kq + (long long)bh * NPAD * HD;
    v8i a = wmma_load_a_i8(qbase, HD);
    for (int ct = 0; ct < CTILES; ++ct) {
        v8i b = wmma_load_b_i8(kbase + (long long)ct * 16 * HD, HD);
        v8i acc = wmma_zero();
        acc = wmma_iu8(a, b, acc);
        int col = ct * 16 + nn;
        #pragma unroll
        for (int r = 0; r < 8; ++r)
            S[half * 8 + r][col] = (float)acc[r] * sqk;
    }
    __syncthreads();

    int row = lane >> 1, part = lane & 1;
    float mx = -3.0e38f;
    for (int c = part; c < SEQ; c += 2) mx = fmaxf(mx, S[row][c]);
    mx = fmaxf(mx, __shfl_xor(mx, 1, 32));
    float sum = 0.0f;
    for (int c = part; c < SEQ; c += 2) sum += expf(S[row][c] - mx);
    sum += __shfl_xor(sum, 1, 32);

    int grow = rt * 16 + row;
    if (part == 0 && grow < SEQ) {
        long long o = (long long)bh * NPAD + grow;
        rowmax[o] = mx;
        rowsum[o] = sum;
        atomicMin(zmin, __float_as_uint(sum));
    }
}

__global__ void attn_pass2_kernel(const int8_t* __restrict__ qq,
                                  const int8_t* __restrict__ kq,
                                  const int8_t* __restrict__ vt,
                                  const float*  __restrict__ scal,
                                  const unsigned* __restrict__ zmin,
                                  const float* __restrict__ rowmax,
                                  const float* __restrict__ rowsum,
                                  unsigned* __restrict__ amax_score,
                                  float* __restrict__ score) {
    __shared__ float  S[16][NPAD];
    __shared__ int8_t P[16][KPAD];
    __shared__ float  rm[16], rs[16];

    int rt = blockIdx.x, bh = blockIdx.y;
    int lane = threadIdx.x & 31, half = lane >> 4, nn = lane & 15;
    float sqk = mkscale(scal[3]) * mkscale(scal[4]);

    const int8_t* qbase = qq + (long long)bh * NPAD * HD + (long long)rt * 16 * HD;
    const int8_t* kbase = kq + (long long)bh * NPAD * HD;
    v8i a = wmma_load_a_i8(qbase, HD);
    for (int ct = 0; ct < CTILES; ++ct) {
        v8i b = wmma_load_b_i8(kbase + (long long)ct * 16 * HD, HD);
        v8i acc = wmma_zero();
        acc = wmma_iu8(a, b, acc);
        int col = ct * 16 + nn;
        #pragma unroll
        for (int r = 0; r < 8; ++r)
            S[half * 8 + r][col] = (float)acc[r] * sqk;
    }
    if (lane < 16) {
        long long o = (long long)bh * NPAD + rt * 16 + lane;
        rm[lane] = rowmax[o];
        rs[lane] = rowsum[o];
    }
    __syncthreads();

    float pmax  = 1.0f / __uint_as_float(*zmin);
    float sattn = fmaxf(pmax * (1.0f / 127.0f), 1e-8f);
    float inv_sattn = 1.0f / sattn;

    for (int idx = lane; idx < 16 * KPAD; idx += 32) {
        int r = idx / KPAD, c = idx % KPAD;
        int8_t qv = 0;
        if (c < SEQ) {
            float p = expf(S[r][c] - rm[r]) / rs[r];
            qv = q8(p, inv_sattn);
        }
        P[r][c] = qv;
    }
    __syncthreads();

    float spv = sattn * mkscale(scal[5]);
    const int8_t* vbase = vt + (long long)bh * HD * KPAD;
    v8i acc[4] = { wmma_zero(), wmma_zero(), wmma_zero(), wmma_zero() };
    for (int kt = 0; kt < KPAD; kt += 64) {
        v8i av = wmma_load_a_i8((const int8_t*)&P[0][0] + kt, KPAD);
        #pragma unroll
        for (int nt = 0; nt < 4; ++nt) {
            v8i bv = wmma_load_b_i8(vbase + (long long)nt * 16 * KPAD + kt, KPAD);
            acc[nt] = wmma_iu8(av, bv, acc[nt]);
        }
    }

    int b_ = bh / HEADS, h = bh % HEADS;
    float lmax = 0.0f;
    #pragma unroll
    for (int nt = 0; nt < 4; ++nt) {
        int col = nt * 16 + nn;
        #pragma unroll
        for (int r = 0; r < 8; ++r) {
            int row = rt * 16 + half * 8 + r;
            if (row < SEQ) {
                float f = (float)acc[nt][r] * spv;
                score[((long long)b_ * SEQ + row) * CDIM + h * HD + col] = f;
                lmax = fmaxf(lmax, fabsf(f));
            }
        }
    }
    lmax = wave_max32(lmax);
    if (lane == 0) atomicMax(amax_score, __float_as_uint(lmax));
}

__global__ void gemm_proj_kernel(const int8_t* __restrict__ sq,
                                 const int8_t* __restrict__ wp,
                                 const float*  __restrict__ bias,
                                 const float*  __restrict__ scal,
                                 float* __restrict__ out) {
    __shared__ int8_t Ws[64 * CDIM];

    int g  = blockIdx.x;
    int wv = threadIdx.x >> 5;
    int mtile = blockIdx.y * 4 + wv;
    int lane = threadIdx.x & 31, half = lane >> 4, nn = lane & 15;

    const int8_t* bstrip = wp + (long long)g * 64 * CDIM;
    lds_strip_load(Ws + wv * 16 * CDIM, bstrip + (long long)wv * 16 * CDIM,
                   16 * CDIM);
    __syncthreads();

    bool valid = (mtile < MTILES);
    int  mt    = valid ? mtile : (MTILES - 1);

    float ssw = mkscale(scal[6]) * mkscale(scal[2]);
    const int8_t* abase = sq + (long long)mt * 16 * CDIM;

    v8i acc[4] = { wmma_zero(), wmma_zero(), wmma_zero(), wmma_zero() };
    for (int kb = 0; kb < CDIM; kb += 64) {
        v8i a = wmma_load_a_i8(abase + kb, CDIM);
        #pragma unroll
        for (int nt = 0; nt < 4; ++nt) {
            v8i b = wmma_load_b_i8(Ws + nt * 16 * CDIM + kb, CDIM);
            acc[nt] = wmma_iu8(a, b, acc[nt]);
        }
    }

    if (valid) {
        #pragma unroll
        for (int nt = 0; nt < 4; ++nt) {
            int oc = g * 64 + nt * 16 + nn;
            float bv = bias[oc];
            #pragma unroll
            for (int r = 0; r < 8; ++r) {
                int mrow = mt * 16 + half * 8 + r;
                out[(long long)mrow * CDIM + oc] = (float)acc[nt][r] * ssw + bv;
            }
        }
    }
}


extern "C" void kernel_launch(void* const* d_in, const int* in_sizes, int n_in,
                              void* d_out, int out_size, void* d_ws, size_t ws_size,
                              hipStream_t stream) {
    const float* x      = (const float*)d_in[0];
    const float* qkv_w  = (const float*)d_in[1];
    const float* qkv_b  = (const float*)d_in[2];
    const float* proj_w = (const float*)d_in[3];
    const float* proj_b = (const float*)d_in[4];
    float* out = (float*)d_out;

    char* w = (char*)d_ws;
    size_t off = 0;
    auto take = [&](size_t bytes) -> void* {
        void* p = w + off;
        off += (bytes + 255) & ~(size_t)255;
        return p;
    };
    unsigned* scal = (unsigned*)take(8 * sizeof(unsigned));
    int8_t* xq   = (int8_t*)take((size_t)MROWS * CDIM);
    int8_t* wq   = (int8_t*)take((size_t)3 * CDIM * CDIM);
    int8_t* wp   = (int8_t*)take((size_t)CDIM * CDIM);
    float*  qf   = (float*) take((size_t)BH * NPAD * HD * 4);
    float*  kf   = (float*) take((size_t)BH * NPAD * HD * 4);
    float*  vf   = (float*) take((size_t)BH * NPAD * HD * 4);
    int8_t* qqb  = (int8_t*)take((size_t)BH * NPAD * HD);
    int8_t* kqb  = (int8_t*)take((size_t)BH * NPAD * HD);
    int8_t* vtb  = (int8_t*)take((size_t)BH * HD * KPAD);
    float*  rmax = (float*) take((size_t)BH * NPAD * 4);
    float*  rsum = (float*) take((size_t)BH * NPAD * 4);
    float*  scoref = (float*)take((size_t)MROWS * CDIM * 4);
    int8_t* sq8  = (int8_t*)take((size_t)MROWS * CDIM);
    (void)ws_size; (void)n_in; (void)in_sizes; (void)out_size;

    const float* scalf = (const float*)scal;

    init_scalars_kernel<<<1, 32, 0, stream>>>(scal);

    absmax_kernel<<<1024, 256, 0, stream>>>(x,      (long long)MROWS * CDIM,  scal + 0);
    absmax_kernel<<<512,  256, 0, stream>>>(qkv_w,  (long long)3 * CDIM * CDIM, scal + 1);
    absmax_kernel<<<256,  256, 0, stream>>>(proj_w, (long long)CDIM * CDIM,   scal + 2);

    quantize_kernel<<<2048, 256, 0, stream>>>(x,      xq, (long long)MROWS * CDIM,  scalf + 0);
    quantize_kernel<<<1024, 256, 0, stream>>>(qkv_w,  wq, (long long)3 * CDIM * CDIM, scalf + 1);
    quantize_kernel<<<512,  256, 0, stream>>>(proj_w, wp, (long long)CDIM * CDIM,   scalf + 2);

    gemm_qkv_kernel<<<dim3(36, 145), 128, 0, stream>>>(
        xq, wq, qkv_b, scalf, scal + 3, qf, kf, vf);

    quantize_qk_kernel<<<2048, 256, 0, stream>>>(qf, qqb, scalf + 3);
    quantize_qk_kernel<<<2048, 256, 0, stream>>>(kf, kqb, scalf + 4);
    quantize_vT_kernel<<<2048, 256, 0, stream>>>(vf, vtb, scalf + 5);

    attn_pass1_kernel<<<dim3(CTILES, BH), 32, 0, stream>>>(
        qqb, kqb, scalf, scal + 7, rmax, rsum);
    attn_pass2_kernel<<<dim3(CTILES, BH), 32, 0, stream>>>(
        qqb, kqb, vtb, scalf, scal + 7, rmax, rsum, scal + 6, scoref);

    quantize_kernel<<<2048, 256, 0, stream>>>(scoref, sq8, (long long)MROWS * CDIM, scalf + 6);

    gemm_proj_kernel<<<dim3(12, 145), 128, 0, stream>>>(sq8, wp, proj_b, scalf, out);
}
